// VectorQuantizer_23545010717034
// MI455X (gfx1250) — matrix operand order repaired and checked by reading, not compiled, not run on hardware
//
#include <hip/hip_runtime.h>
#include <hip/hip_bf16.h>
#include <math.h>

#define BATCH   16384
#define EMB_DIM 512
#define NUM_EMB 4096
#define MTILES  2
#define NT      2            // N-tiles (codes per chunk = NT*16)
#define SROW    520          // padded LDS row stride in elements (1040B: conflict-free b128 reads)
#define NCHUNK  (NUM_EMB / (NT * 16))

typedef __attribute__((ext_vector_type(16))) __bf16 v16bf;
typedef __attribute__((ext_vector_type(8)))  __bf16 v8bf;
typedef __attribute__((ext_vector_type(8)))  float  v8f;

// ---------------- CDNA5 async global->LDS copy helpers ----------------
__device__ __forceinline__ void async_b128(unsigned lds_addr, const __bf16* g) {
  // GLOBAL_LOAD_ASYNC_TO_LDS_B128: VDST = LDS byte address, VADDR = 64-bit global address.
  // Per-lane: LDS[lds_addr] = MEM[g], 16 bytes. Tracked by ASYNCcnt.
  asm volatile("global_load_async_to_lds_b128 %0, %1, off"
               :: "v"(lds_addr), "v"((unsigned long long)(uintptr_t)g)
               : "memory");
}
__device__ __forceinline__ void wait_async_le32() {
  asm volatile("s_wait_asynccnt 0x20" ::: "memory");
}
__device__ __forceinline__ void wait_async_0() {
  asm volatile("s_wait_asynccnt 0x0" ::: "memory");
}

// ---------------- prep: split z into bf16 hi/lo ----------------
__global__ void vq_prep_z(const float* __restrict__ z,
                          __bf16* __restrict__ zhi, __bf16* __restrict__ zlo, int n) {
  int i = blockIdx.x * blockDim.x + threadIdx.x;
  if (i >= n) return;
  float x = z[i];
  __bf16 h = (__bf16)x;
  zhi[i] = h;
  zlo[i] = (__bf16)(x - (float)h);
}

// ---------------- prep: split emb into hi/lo + row norms ----------------
__global__ void vq_prep_e(const float* __restrict__ e,
                          __bf16* __restrict__ ehi, __bf16* __restrict__ elo,
                          float* __restrict__ enorm2) {
  int lane = threadIdx.x & 31;
  int row  = (blockIdx.x * blockDim.x + threadIdx.x) >> 5;
  if (row >= NUM_EMB) return;
  const float* er = e + (size_t)row * EMB_DIM;
  float sq = 0.f;
#pragma unroll
  for (int t = 0; t < EMB_DIM / 32; ++t) {
    int i = lane + 32 * t;
    float x = er[i];
    __bf16 h = (__bf16)x;
    ehi[(size_t)row * EMB_DIM + i] = h;
    elo[(size_t)row * EMB_DIM + i] = (__bf16)(x - (float)h);
    sq += x * x;
  }
#pragma unroll
  for (int m = 16; m >= 1; m >>= 1) sq += __shfl_xor(sq, m, 32);
  if (lane == 0) enorm2[row] = sq;
}

// ---------------- init scratch accumulators ----------------
__global__ void vq_init(float* __restrict__ counts, float* __restrict__ sse) {
  int i = blockIdx.x * blockDim.x + threadIdx.x;
  if (i < NUM_EMB) counts[i] = 0.f;
  if (i == 0) *sse = 0.f;
}

// ---------------- fused GEMM + argmin (WMMA bf16 hi/lo, LDS-staged B) ----------------
// dist[m,k] = ||e_k||^2 - 2 * dot(z_m, e_k); argmin over k.
// 4 waves/block share each 32-code chunk via async global->LDS double buffering.
__global__ void __launch_bounds__(128)
vq_argmin(const __bf16* __restrict__ zhi, const __bf16* __restrict__ zlo,
          const __bf16* __restrict__ ehi, const __bf16* __restrict__ elo,
          const float* __restrict__ enorm2, int* __restrict__ idx_out) {
  // LDS: [buffer 0/1][hi/lo][NT*16 code rows][SROW padded elems]  (~130KB)
  __shared__ __attribute__((aligned(16))) __bf16 sh[2 * 2 * (NT * 16) * SROW];

  const int tid   = threadIdx.x;
  const int lane  = tid & 31;
  const int wave  = tid >> 5;
  const int hl    = lane >> 4;      // wave half
  const int l15   = lane & 15;
  // A operand (16x32 bf16): lanes 0-15 hold K{0..7,16..23}, lanes 16-31 K{8..15,24..31}
  const int aoff  = hl ? 8 : 0;
  // B operand (32x16 bf16): lanes 0-15 hold K{0..7,16..23}, lanes 16-31 K{8..15,24..31}
  const int boff  = hl ? 8 : 0;

  const int m0 = (blockIdx.x * 4 + wave) * (16 * MTILES);

  const __bf16* arow_h[MTILES];
  const __bf16* arow_l[MTILES];
#pragma unroll
  for (int t = 0; t < MTILES; ++t) {
    size_t r = (size_t)(m0 + t * 16 + l15) * EMB_DIM;
    arow_h[t] = zhi + r;
    arow_l[t] = zlo + r;
  }

  // copy role: 4 threads per code row (32 rows), each moves 256B of hi and 256B of lo
  const int crow = tid >> 2;        // 0..31
  const int cseg = tid & 3;         // 0..3

  float best[MTILES][8];
  int   bidx[MTILES][8];
#pragma unroll
  for (int t = 0; t < MTILES; ++t)
#pragma unroll
    for (int j = 0; j < 8; ++j) { best[t][j] = 3.4e38f; bidx[t][j] = 0; }

  // ---- issue copies for chunk 0 into buffer 0 ----
  {
    const __bf16* gh = ehi + (size_t)crow * EMB_DIM + cseg * 128;
    const __bf16* gl = elo + (size_t)crow * EMB_DIM + cseg * 128;
    unsigned lh = (unsigned)(uintptr_t)&sh[((0 * 2 + 0) * (NT * 16) + crow) * SROW + cseg * 128];
    unsigned ll = (unsigned)(uintptr_t)&sh[((0 * 2 + 1) * (NT * 16) + crow) * SROW + cseg * 128];
#pragma unroll
    for (int i = 0; i < 16; ++i) {
      async_b128(lh + i * 16, gh + i * 8);
      async_b128(ll + i * 16, gl + i * 8);
    }
  }

  for (int c = 0; c < NCHUNK; ++c) {
    const int cur = c & 1;
    if (c + 1 < NCHUNK) {
      // prefetch next chunk into the other buffer
      const int nb = cur ^ 1;
      const __bf16* gh = ehi + (size_t)((c + 1) * (NT * 16) + crow) * EMB_DIM + cseg * 128;
      const __bf16* gl = elo + (size_t)((c + 1) * (NT * 16) + crow) * EMB_DIM + cseg * 128;
      unsigned lh = (unsigned)(uintptr_t)&sh[((nb * 2 + 0) * (NT * 16) + crow) * SROW + cseg * 128];
      unsigned ll = (unsigned)(uintptr_t)&sh[((nb * 2 + 1) * (NT * 16) + crow) * SROW + cseg * 128];
#pragma unroll
      for (int i = 0; i < 16; ++i) {
        async_b128(lh + i * 16, gh + i * 8);
        async_b128(ll + i * 16, gl + i * 8);
      }
      wait_async_le32();   // the 32 older copies (current buffer) have landed
    } else {
      wait_async_0();
    }
    __syncthreads();       // all waves' copies for `cur` visible

    int bh_base[NT], bl_base[NT];
#pragma unroll
    for (int nt = 0; nt < NT; ++nt) {
      bh_base[nt] = ((cur * 2 + 0) * (NT * 16) + nt * 16 + l15) * SROW + boff;
      bl_base[nt] = ((cur * 2 + 1) * (NT * 16) + nt * 16 + l15) * SROW + boff;
    }

    v8f acc[MTILES][NT] = {};

#pragma unroll 2
    for (int kb = 0; kb < EMB_DIM; kb += 32) {
      v16bf Ah[MTILES], Al[MTILES];
#pragma unroll
      for (int t = 0; t < MTILES; ++t) {
        v8bf a0 = *(const v8bf*)(arow_h[t] + kb + aoff);
        v8bf a1 = *(const v8bf*)(arow_h[t] + kb + aoff + 16);
        Ah[t] = __builtin_shufflevector(a0, a1, 0,1,2,3,4,5,6,7,8,9,10,11,12,13,14,15);
        v8bf l0 = *(const v8bf*)(arow_l[t] + kb + aoff);
        v8bf l1 = *(const v8bf*)(arow_l[t] + kb + aoff + 16);
        Al[t] = __builtin_shufflevector(l0, l1, 0,1,2,3,4,5,6,7,8,9,10,11,12,13,14,15);
      }
#pragma unroll
      for (int nt = 0; nt < NT; ++nt) {
        v8bf b0 = *(const v8bf*)&sh[bh_base[nt] + kb];
        v8bf b1 = *(const v8bf*)&sh[bh_base[nt] + kb + 16];
        v16bf Bh = __builtin_shufflevector(b0, b1, 0,1,2,3,4,5,6,7,8,9,10,11,12,13,14,15);
        v8bf c0v = *(const v8bf*)&sh[bl_base[nt] + kb];
        v8bf c1v = *(const v8bf*)&sh[bl_base[nt] + kb + 16];
        v16bf Bl = __builtin_shufflevector(c0v, c1v, 0,1,2,3,4,5,6,7,8,9,10,11,12,13,14,15);

#pragma unroll
        for (int t = 0; t < MTILES; ++t) {
          acc[t][nt] = __builtin_amdgcn_wmma_f32_16x16x32_bf16(false, Ah[t], false, Bh, (short)0, acc[t][nt], false, false);
          acc[t][nt] = __builtin_amdgcn_wmma_f32_16x16x32_bf16(false, Ah[t], false, Bl, (short)0, acc[t][nt], false, false);
          acc[t][nt] = __builtin_amdgcn_wmma_f32_16x16x32_bf16(false, Al[t], false, Bh, (short)0, acc[t][nt], false, false);
        }
      }
    }

#pragma unroll
    for (int nt = 0; nt < NT; ++nt) {
      const int  code = c * (NT * 16) + nt * 16 + l15;
      const float en  = enorm2[code];
#pragma unroll
      for (int t = 0; t < MTILES; ++t)
#pragma unroll
        for (int j = 0; j < 8; ++j) {
          float d = en - 2.f * acc[t][nt][j];
          if (d < best[t][j]) { best[t][j] = d; bidx[t][j] = code; }
        }
    }

    __syncthreads();       // readers done before this buffer is refilled (chunk c+2)
  }

  // Butterfly min-reduce across the 16 columns (lanes) in each wave half.
#pragma unroll
  for (int m = 1; m <= 8; m <<= 1) {
#pragma unroll
    for (int t = 0; t < MTILES; ++t)
#pragma unroll
      for (int j = 0; j < 8; ++j) {
        float ob = __shfl_xor(best[t][j], m, 32);
        int   oi = __shfl_xor(bidx[t][j], m, 32);
        if (ob < best[t][j] || (ob == best[t][j] && oi < bidx[t][j])) {
          best[t][j] = ob; bidx[t][j] = oi;
        }
      }
  }

  if (l15 == 0) {
    // C layout: VGPR j holds row M = j + 8*hl of the tile.
#pragma unroll
    for (int t = 0; t < MTILES; ++t)
#pragma unroll
      for (int j = 0; j < 8; ++j)
        idx_out[m0 + t * 16 + hl * 8 + j] = bidx[t][j];
  }
}

// ---------------- Householder rotation + outputs (one wave per row) ----------------
__global__ void __launch_bounds__(256)
vq_house(const float* __restrict__ z, const float* __restrict__ emb,
         const int* __restrict__ idx, float* __restrict__ qout,
         float* __restrict__ out_idx_f, float* __restrict__ sse,
         float* __restrict__ counts) {
  int lane = threadIdx.x & 31;
  int row  = (blockIdx.x * blockDim.x + threadIdx.x) >> 5;
  if (row >= BATCH) return;
  int k = idx[row];
  const float* zr = z   + (size_t)row * EMB_DIM;
  const float* er = emb + (size_t)k   * EMB_DIM;

  float zv[16], ev[16];
  float zz = 0.f, ee = 0.f;
#pragma unroll
  for (int t = 0; t < 16; ++t) {
    int i = lane + 32 * t;
    zv[t] = zr[i]; ev[t] = er[i];
    zz += zv[t] * zv[t]; ee += ev[t] * ev[t];
  }
#pragma unroll
  for (int m = 16; m >= 1; m >>= 1) { zz += __shfl_xor(zz, m, 32); ee += __shfl_xor(ee, m, 32); }

  float rnz = 1.f / fmaxf(sqrtf(zz), 1e-12f);
  float rne = 1.f / fmaxf(sqrtf(ee), 1e-12f);

  float vv = 0.f, vz = 0.f;
#pragma unroll
  for (int t = 0; t < 16; ++t) {
    float v = zv[t] * rnz - ev[t] * rne;
    vv += v * v; vz += v * zv[t];
  }
#pragma unroll
  for (int m = 16; m >= 1; m >>= 1) { vv += __shfl_xor(vv, m, 32); vz += __shfl_xor(vz, m, 32); }

  float vn = sqrtf(vv);
  float s  = (vn > 1e-5f) ? 1.f / (vn + 1e-8f) : 1.f;
  float coef = 2.f * s * s * vz;   // q = z - 2*(s*v)*(s*v . z)

  float le = 0.f;
#pragma unroll
  for (int t = 0; t < 16; ++t) {
    int i = lane + 32 * t;
    float v = zv[t] * rnz - ev[t] * rne;
    float q = zv[t] - coef * v;
    qout[(size_t)row * EMB_DIM + i] = q;   // q_st == q_out numerically
    float d = q - zv[t];
    le += d * d;
  }
#pragma unroll
  for (int m = 16; m >= 1; m >>= 1) le += __shfl_xor(le, m, 32);
  if (lane == 0) {
    atomicAdd(sse, le);
    atomicAdd(&counts[k], 1.f);
    out_idx_f[row] = (float)k;
  }
}

// ---------------- finalize: loss scalar + perplexity ----------------
__global__ void vq_final(const float* __restrict__ counts, const float* __restrict__ sse,
                         float* __restrict__ out_loss, float* __restrict__ out_ppl) {
  __shared__ float sm[256];
  int tid = threadIdx.x;
  float p = 0.f;
  for (int k = tid; k < NUM_EMB; k += 256) {
    float pr = counts[k] * (1.f / (float)BATCH);
    p += pr * logf(pr + 1e-10f);
  }
  sm[tid] = p; __syncthreads();
  for (int s = 128; s > 0; s >>= 1) { if (tid < s) sm[tid] += sm[tid + s]; __syncthreads(); }
  if (tid == 0) {
    // q_latent == e_latent numerically through stop_gradient => loss = (1 + 0.25) * mse
    *out_loss = 1.25f * (*sse) * (1.f / ((float)BATCH * (float)EMB_DIM));
    *out_ppl  = expf(-sm[0]);
  }
}

extern "C" void kernel_launch(void* const* d_in, const int* in_sizes, int n_in,
                              void* d_out, int out_size, void* d_ws, size_t ws_size,
                              hipStream_t stream) {
  const float* z   = (const float*)d_in[0];
  const float* emb = (const float*)d_in[1];
  float* out = (float*)d_out;

  char* ws = (char*)d_ws;
  const size_t BD = (size_t)BATCH * EMB_DIM;       // 8388608
  const size_t KD = (size_t)NUM_EMB * EMB_DIM;     // 2097152
  __bf16* zhi    = (__bf16*)(ws);
  __bf16* zlo    = (__bf16*)(ws + BD * 2);
  __bf16* ehi    = (__bf16*)(ws + BD * 4);
  __bf16* elo    = (__bf16*)(ws + BD * 4 + KD * 2);
  float*  enorm2 = (float*) (ws + BD * 4 + KD * 4);
  int*    idxbuf = (int*)   (ws + BD * 4 + KD * 4 + NUM_EMB * 4);
  float*  counts = (float*) (ws + BD * 4 + KD * 4 + NUM_EMB * 4 + BATCH * 4);
  float*  ssebuf = (float*) (ws + BD * 4 + KD * 4 + NUM_EMB * 4 + BATCH * 4 + NUM_EMB * 4);

  float* q_st   = out;             // [BATCH*EMB_DIM]
  float* loss   = out + BD;        // [1]
  float* ppl    = out + BD + 1;    // [1]
  float* idx_f  = out + BD + 2;    // [BATCH]

  vq_prep_z<<<(int)((BD + 255) / 256), 256, 0, stream>>>(z, zhi, zlo, (int)BD);
  vq_prep_e<<<NUM_EMB / 8, 256, 0, stream>>>(emb, ehi, elo, enorm2);
  vq_init<<<(NUM_EMB + 255) / 256, 256, 0, stream>>>(counts, ssebuf);

  // 128 blocks * 4 waves * (2 tiles * 16 rows) = 16384 rows
  vq_argmin<<<128, 128, 0, stream>>>(zhi, zlo, ehi, elo, enorm2, idxbuf);

  vq_house<<<BATCH / 8, 256, 0, stream>>>(z, emb, idxbuf, q_st, idx_f, ssebuf, counts);
  vq_final<<<1, 256, 0, stream>>>(counts, ssebuf, loss, ppl);
}
